// CAMEncoder_87600152969326
// MI455X (gfx1250) — matrix operand order repaired and checked by reading, not compiled, not run on hardware
//
#include <hip/hip_runtime.h>
#include <hip/hip_bf16.h>

// ---------------------------------------------------------------------------
// GATv2 encoder for MI455X (gfx1250, wave32).
// Dense node transforms -> v_wmma_f32_16x16x32_bf16 (bf16 in, f32 acc).
// Edge softmax/aggregation -> wave-per-edge coalesced gathers + f32 atomics.
// ---------------------------------------------------------------------------

#define NND 100000
#define NE  800000
#define SLOPE_F 0.15f

typedef __attribute__((ext_vector_type(16))) __bf16 v16bf;
typedef __attribute__((ext_vector_type(8)))  float  v8f;

union Frag16 {
    v16bf v;
    unsigned short u[16];
    uint4 q[2];
};

__device__ __forceinline__ unsigned short f2bf(float f) {
    unsigned int u = __float_as_uint(f);
    u += 0x7FFFu + ((u >> 16) & 1u);          // round-to-nearest-even
    return (unsigned short)(u >> 16);
}

__device__ __forceinline__ void atomicMaxF(float* addr, float val) {
    if (val >= 0.f) atomicMax((int*)addr, __float_as_int(val));
    else            atomicMin((unsigned int*)addr, __float_as_uint(val));
}

// -------------------------------- utility kernels ---------------------------

__global__ void fill_f32(float* __restrict__ p, float v, long long n) {
    long long i = blockIdx.x * (long long)blockDim.x + threadIdx.x;
    if (i < n) p[i] = v;
}

__global__ void cvt_f32_bf16(const float* __restrict__ s, unsigned short* __restrict__ d,
                             long long n) {
    long long i = blockIdx.x * (long long)blockDim.x + threadIdx.x;
    if (i < n) d[i] = f2bf(s[i]);
}

// copy x (N x 64) into column block [0,64) of dst with row stride ldd
__global__ void copy_x_cols(const float* __restrict__ x, float* __restrict__ dst, int ldd) {
    long long i = blockIdx.x * (long long)blockDim.x + threadIdx.x;
    if (i >= (long long)NND * 64) return;
    int n = (int)(i >> 6);
    int c = (int)(i & 63);
    dst[(size_t)n * ldd + c] = x[i];
}

// Pack bf16 weight matrix W[K x N] (row major) into per-wave WMMA B fragments:
// layout [kBlock][nTile][lane(32)][16 ushort]. Per lane: n = nT*16 + (lane&15),
// element i covers K = kb*32 + (lane>>4)*16 + i.
__global__ void pack_b_kernel(const unsigned short* __restrict__ W,
                              unsigned short* __restrict__ Bp, int K, int N) {
    int idx = blockIdx.x * blockDim.x + threadIdx.x;
    int nTiles = N >> 4;
    int total = (K >> 5) * nTiles * 32;
    if (idx >= total) return;
    int lane = idx & 31;
    int t = idx >> 5;
    int nT = t % nTiles;
    int kb = t / nTiles;
    int n  = nT * 16 + (lane & 15);
    int k0 = kb * 32 + (lane >> 4) * 16;
    unsigned short* out = Bp + (size_t)idx * 16;
#pragma unroll
    for (int i = 0; i < 16; ++i)
        out[i] = W[(size_t)(k0 + i) * N + n];
}

// -------------------------------- WMMA GEMM ---------------------------------
// C[M x N] = A[M x K](bf16,row major, stride lda) @ Bpacked + bias, optional ReLU,
// written to C with row stride ldc at column offset colOff. One wave per 16x16 tile.
__global__ void gemm_bf16_wmma(const unsigned short* __restrict__ A,
                               const unsigned short* __restrict__ Bp,
                               const float* __restrict__ bias,
                               float* __restrict__ C,
                               int M, int N, int K, int lda, int ldc,
                               int colOff, int doRelu) {
    const int lane = threadIdx.x & 31;
    const int wid  = (blockIdx.x * blockDim.x + threadIdx.x) >> 5;
    const int nTiles = N >> 4;
    const int tiles  = (M >> 4) * nTiles;
    if (wid >= tiles) return;
    const int mT = wid / nTiles;
    const int nT = wid - mT * nTiles;
    const int kh = lane >> 4;     // K-half select
    const int ml = lane & 15;     // row (A) / col (B,D) within tile

    const unsigned short* arow = A + (size_t)(mT * 16 + ml) * lda;
    const uint4* bfr = (const uint4*)Bp + ((size_t)nT * 32 + lane) * 2;
    const size_t bstep = (size_t)nTiles * 64;   // uint4 per K-block of packed B

    v8f acc = {};
    const int kBlocks = K >> 5;
    for (int kb = 0; kb < kBlocks; ++kb) {
        Frag16 a, b;
        a.q[0] = *(const uint4*)(arow + kb * 32 + kh * 16);
        a.q[1] = *(const uint4*)(arow + kb * 32 + kh * 16 + 8);
        b.q[0] = bfr[0];
        b.q[1] = bfr[1];
        bfr += bstep;
        if (kb + 1 < kBlocks)
            __builtin_prefetch(arow + (kb + 1) * 32, 0, 3);   // global_prefetch_b8
        acc = __builtin_amdgcn_wmma_f32_16x16x32_bf16(
            false, a.v, false, b.v, (short)0, acc, false, false);
    }

    const float bv = bias[nT * 16 + ml];
    float* crow = C + (size_t)(mT * 16 + kh * 8) * ldc + colOff + nT * 16 + ml;
#pragma unroll
    for (int r = 0; r < 8; ++r) {            // D layout: VGPR r -> row r + 8*kh
        float v = acc[r] + bv;
        if (doRelu) v = v > 0.f ? v : 0.f;
        crow[(size_t)r * ldc] = v;
    }
}

// -------------------------------- edge kernels ------------------------------
// logits[e,h] = sum_c leaky_relu(xl[src,h,c] + xr[dst,h,c]) * att[h,c]
// also segment-max into mmax[dst,h]. One wave per edge, lanes over channels.
__global__ void edge_logits(const float* __restrict__ xl, const float* __restrict__ xr,
                            const float* __restrict__ att, const long long* __restrict__ edges,
                            float* __restrict__ logit, float* __restrict__ mmax, int C) {
    const int lane = threadIdx.x & 31;
    const int e = (blockIdx.x * blockDim.x + threadIdx.x) >> 5;
    if (e >= NE) return;
    const int src = (int)edges[e];
    const int dst = (int)edges[NE + e];
    const int HC = 2 * C;
    const float* xs = xl + (size_t)src * HC;
    const float* xd = xr + (size_t)dst * HC;
#pragma unroll
    for (int h = 0; h < 2; ++h) {
        float sum = 0.f;
        for (int c = lane; c < C; c += 32) {
            float v = xs[h * C + c] + xd[h * C + c];
            v = v > 0.f ? v : v * SLOPE_F;
            sum += v * att[h * C + c];
        }
        for (int off = 16; off > 0; off >>= 1) sum += __shfl_xor(sum, off);
        if (lane == 0) {
            logit[(size_t)e * 2 + h] = sum;
            atomicMaxF(&mmax[(size_t)dst * 2 + h], sum);
        }
    }
}

// p = exp(logit - max[dst]); denom[dst] += p  (overwrites logit with p)
__global__ void edge_expsum(const long long* __restrict__ edges, float* __restrict__ logit,
                            const float* __restrict__ mmax, float* __restrict__ denom) {
    long long i = blockIdx.x * (long long)blockDim.x + threadIdx.x;
    if (i >= (long long)NE * 2) return;
    int e = (int)(i >> 1);
    int h = (int)(i & 1);
    int dst = (int)edges[NE + e];
    float pv = __expf(logit[i] - mmax[dst * 2 + h]);
    logit[i] = pv;
    atomicAdd(&denom[dst * 2 + h], pv);
}

// aggr[dst,c] += (p/denom) * xl[src,h,c] / H    (head mean folded in)
__global__ void edge_aggregate(const long long* __restrict__ edges,
                               const float* __restrict__ logit, const float* __restrict__ denom,
                               const float* __restrict__ xl, float* __restrict__ aggr, int C) {
    const int lane = threadIdx.x & 31;
    const int e = (blockIdx.x * blockDim.x + threadIdx.x) >> 5;
    if (e >= NE) return;
    const int src = (int)edges[e];
    const int dst = (int)edges[NE + e];
    const int HC = 2 * C;
    const float* xs = xl + (size_t)src * HC;
    float* out = aggr + (size_t)dst * C;
#pragma unroll
    for (int h = 0; h < 2; ++h) {
        float a = logit[(size_t)e * 2 + h] / denom[dst * 2 + h] * 0.5f;
        for (int c = lane; c < C; c += 32)
            atomicAdd(&out[c], a * xs[h * C + c]);
    }
}

// dst[n, colOff+c] = relu(aggr[n,c] + bias[c])
__global__ void finalize_gat(const float* __restrict__ aggr, const float* __restrict__ bo,
                             float* __restrict__ dst, int C, int ldd, int colOff) {
    long long i = blockIdx.x * (long long)blockDim.x + threadIdx.x;
    if (i >= (long long)NND * C) return;
    int n = (int)(i / C);
    int c = (int)(i - (long long)n * C);
    float v = aggr[i] + bo[c];
    dst[(size_t)n * ldd + colOff + c] = v > 0.f ? v : 0.f;
}

// -------------------------------- host orchestration ------------------------

static inline int cdiv(long long a, long long b) { return (int)((a + b - 1) / b); }

static void gemm_full(hipStream_t s, const unsigned short* A, const float* W,
                      const float* bias, unsigned short* wbf, unsigned short* wpk,
                      float* C, int M, int N, int K, int lda, int ldc,
                      int colOff, int doRelu) {
    long long wn = (long long)K * N;
    cvt_f32_bf16<<<cdiv(wn, 256), 256, 0, s>>>(W, wbf, wn);
    int pt = (K / 32) * (N / 16) * 32;
    pack_b_kernel<<<cdiv(pt, 256), 256, 0, s>>>(wbf, wpk, K, N);
    long long th = (long long)(M / 16) * (N / 16) * 32;
    gemm_bf16_wmma<<<cdiv(th, 256), 256, 0, s>>>(A, wpk, bias, C, M, N, K, lda, ldc,
                                                 colOff, doRelu);
}

static void run_gat(hipStream_t s, const unsigned short* Abf, int K,
                    const float* Wl, const float* bl, const float* Wr, const float* br,
                    const float* att, const float* bo,
                    const long long* edges, int C,
                    float* dst, int ldd, int colOff,
                    float* xl, float* xr, float* logit, float* mmax, float* denom,
                    float* aggr, unsigned short* wbf, unsigned short* wpk) {
    const int HC = 2 * C;
    gemm_full(s, Abf, Wl, bl, wbf, wpk, xl, NND, HC, K, K, HC, 0, 0);
    gemm_full(s, Abf, Wr, br, wbf, wpk, xr, NND, HC, K, K, HC, 0, 0);
    fill_f32<<<cdiv((long long)NND * 2, 256), 256, 0, s>>>(mmax, -__builtin_inff(),
                                                           (long long)NND * 2);
    fill_f32<<<cdiv((long long)NND * 2, 256), 256, 0, s>>>(denom, 0.f, (long long)NND * 2);
    fill_f32<<<cdiv((long long)NND * C, 256), 256, 0, s>>>(aggr, 0.f, (long long)NND * C);
    edge_logits<<<cdiv((long long)NE * 32, 256), 256, 0, s>>>(xl, xr, att, edges, logit,
                                                              mmax, C);
    edge_expsum<<<cdiv((long long)NE * 2, 256), 256, 0, s>>>(edges, logit, mmax, denom);
    edge_aggregate<<<cdiv((long long)NE * 32, 256), 256, 0, s>>>(edges, logit, denom, xl,
                                                                 aggr, C);
    finalize_gat<<<cdiv((long long)NND * C, 256), 256, 0, s>>>(aggr, bo, dst, C, ldd, colOff);
}

extern "C" void kernel_launch(void* const* d_in, const int* in_sizes, int n_in,
                              void* d_out, int out_size, void* d_ws, size_t ws_size,
                              hipStream_t stream) {
    (void)in_sizes; (void)n_in; (void)out_size; (void)ws_size;
    const float*     x     = (const float*)d_in[0];
    const long long* e_ops = (const long long*)d_in[1];
    const long long* e_job = (const long long*)d_in[2];
    const long long* e_mac = (const long long*)d_in[3];
    auto F = [&](int i) { return (const float*)d_in[i]; };
    const float* linW = F(40);
    const float* linB = F(41);
    float* out = (float*)d_out;

    // ---- carve workspace (~570 MB) ----
    char* p = (char*)d_ws;
    auto take = [&](size_t bytes) -> void* {
        void* r = (void*)p;
        p += (bytes + 255) & ~(size_t)255;
        return r;
    };
    float* hcat  = (float*)take((size_t)NND * 256 * 4);   // [x | h1_job | h1_mac | h1_ops]
    float* h2cat = (float*)take((size_t)NND * 384 * 4);   // [h2_ops | h2_job | h2_mac]
    float* xl    = (float*)take((size_t)NND * 256 * 4);
    float* xr    = (float*)take((size_t)NND * 256 * 4);
    unsigned short* hbf = (unsigned short*)take((size_t)NND * 256 * 2);
    float* logit = (float*)take((size_t)NE * 2 * 4);
    float* mmax  = (float*)take((size_t)NND * 2 * 4);
    float* denom = (float*)take((size_t)NND * 2 * 4);
    float* aggr  = (float*)take((size_t)NND * 128 * 4);
    unsigned short* wbf = (unsigned short*)take(65536 * 2);
    unsigned short* wpk = (unsigned short*)take(65536 * 2);
    unsigned short* xbf  = hbf;                 // alias: only live during layer 1
    unsigned short* h2bf = (unsigned short*)xr; // alias: only live after last conv

    // ---- layer 1 (IN=64 -> H=2 x C=64), outputs into hcat column blocks ----
    cvt_f32_bf16<<<cdiv((long long)NND * 64, 256), 256, 0, stream>>>(x, xbf,
                                                                     (long long)NND * 64);
    run_gat(stream, xbf, 64, F(4),  F(5),  F(6),  F(7),  F(8),  F(9),  e_ops, 64,
            hcat, 256, 192, xl, xr, logit, mmax, denom, aggr, wbf, wpk);  // h1_ops
    run_gat(stream, xbf, 64, F(10), F(11), F(12), F(13), F(14), F(15), e_job, 64,
            hcat, 256, 64,  xl, xr, logit, mmax, denom, aggr, wbf, wpk);  // h1_job
    run_gat(stream, xbf, 64, F(16), F(17), F(18), F(19), F(20), F(21), e_mac, 64,
            hcat, 256, 128, xl, xr, logit, mmax, denom, aggr, wbf, wpk);  // h1_mac
    copy_x_cols<<<cdiv((long long)NND * 64, 256), 256, 0, stream>>>(x, hcat, 256);

    // ---- layer 2 (256 -> H=2 x C=128), outputs into h2cat column blocks ----
    cvt_f32_bf16<<<cdiv((long long)NND * 256, 256), 256, 0, stream>>>(hcat, hbf,
                                                                      (long long)NND * 256);
    run_gat(stream, hbf, 256, F(22), F(23), F(24), F(25), F(26), F(27), e_ops, 128,
            h2cat, 384, 0,   xl, xr, logit, mmax, denom, aggr, wbf, wpk); // h2_ops
    run_gat(stream, hbf, 256, F(28), F(29), F(30), F(31), F(32), F(33), e_job, 128,
            h2cat, 384, 128, xl, xr, logit, mmax, denom, aggr, wbf, wpk); // h2_job
    run_gat(stream, hbf, 256, F(34), F(35), F(36), F(37), F(38), F(39), e_mac, 128,
            h2cat, 384, 256, xl, xr, logit, mmax, denom, aggr, wbf, wpk); // h2_mac

    // ---- final: out[:,64:192] = relu(h2cat @ lin_W + lin_b); out[:,0:64] = x ----
    cvt_f32_bf16<<<cdiv((long long)NND * 384, 256), 256, 0, stream>>>(h2cat, h2bf,
                                                                      (long long)NND * 384);
    gemm_full(stream, h2bf, linW, linB, wbf, wpk, out, NND, 128, 384, 384, 192, 64, 1);
    copy_x_cols<<<cdiv((long long)NND * 64, 256), 256, 0, stream>>>(x, out, 192);
}
